// GATSAGEMultiTaskPredictor_70703751627524
// MI455X (gfx1250) — matrix operand order repaired and checked by reading, not compiled, not run on hardware
//
#include <hip/hip_runtime.h>

typedef __attribute__((ext_vector_type(16))) _Float16 v16h;
typedef __attribute__((ext_vector_type(8)))  _Float16 v8h;
typedef __attribute__((ext_vector_type(2)))  _Float16 v2h;
typedef __attribute__((ext_vector_type(8)))  float    v8f;
typedef __attribute__((ext_vector_type(4)))  int      v4i;

#define AS1 __attribute__((address_space(1)))
#define AS3 __attribute__((address_space(3)))

#if defined(__gfx1250__) && __has_builtin(__builtin_amdgcn_global_load_async_to_lds_b128)
#define USE_ASYNC 1
#else
#define USE_ASYNC 0
#endif

// ---------------------------------------------------------------------------
// Weight pre-pack: f32 [K, Mo] row-major -> f16 WMMA-B fragment order.
// frag index = ((ct*(K/32) + kc)*32 + lane)*16 + j
//   lane: n = ct*16 + lane%16, half = lane/16
//   j (0..15): k = kc*32 + half*16 + j
// ---------------------------------------------------------------------------
__global__ void pack_w(const float* __restrict__ W, _Float16* __restrict__ P,
                       int K, int Mo, int MoPad) {
    int tid = blockIdx.x * blockDim.x + threadIdx.x;
    int KC = K >> 5;
    int total = (MoPad >> 4) * KC * 32;
    if (tid >= total) return;
    int lane = tid & 31;
    int kc   = (tid >> 5) % KC;
    int ct   = tid / (32 * KC);
    int half = lane >> 4;
    int n    = (ct << 4) + (lane & 15);
    _Float16* dst = P + (size_t)tid * 16;
    #pragma unroll
    for (int j = 0; j < 16; ++j) {
        int k = (kc << 5) + (half << 4) + j;
        float v = (n < Mo) ? W[(size_t)k * Mo + n] : 0.0f;
        dst[j] = (_Float16)v;
    }
}

// ---------------------------------------------------------------------------
// WMMA GEMM: C[M,ldc](cols<Mo) = A[M,K] @ Wp (+C if acc) (+bias) (PReLU *pa)
// Block: 256 thr = 8 waves; covers 32 rows x 128 cols.
// A tile: async global->LDS (f32), then cooperative convert to f16 in LDS.
// Each wave: 2 row-strips x 1 col-tile => 2 v_wmma per B-fragment load.
// ---------------------------------------------------------------------------
__global__ void wmma_gemm(const float* __restrict__ A, const _Float16* __restrict__ Wp,
                          float* __restrict__ C, const float* __restrict__ bias,
                          const float* __restrict__ pa,
                          int M, int K, int Mo, int ldc, int acc) {
    extern __shared__ char smem[];
    float*    sF = (float*)smem;                                 // 32 x K f32
    _Float16* sH = (_Float16*)(smem + (size_t)32 * K * 4);       // 32 x K f16
    const int tid = threadIdx.x;
    const int rowbase = blockIdx.x << 5;

    // stage 32 x K f32 tile into LDS (16B chunks per thread)
    const int kq4 = K >> 2;
    const int chunks = kq4 << 5;
    for (int i = tid; i < chunks; i += 256) {
        int r  = i / kq4;
        int kq = i - r * kq4;
        int row = rowbase + r;
        float* lp = sF + r * K + (kq << 2);
        if (row < M) {
            const float* gp = A + (size_t)row * K + (kq << 2);
#if USE_ASYNC
            __builtin_amdgcn_global_load_async_to_lds_b128(
                (AS1 v4i*)gp, (AS3 v4i*)lp, 0, 0);
#else
            *(float4*)lp = *(const float4*)gp;
#endif
        } else {
            float4 z = {0.0f, 0.0f, 0.0f, 0.0f};
            *(float4*)lp = z;
        }
    }
#if USE_ASYNC
#if __has_builtin(__builtin_amdgcn_s_wait_asynccnt)
    __builtin_amdgcn_s_wait_asynccnt(0);
#else
    asm volatile("s_wait_asynccnt 0x0" ::: "memory");
#endif
#endif
    __syncthreads();

    // convert f32 tile -> f16 tile (2 elems / thread / iter)
    const int tot2 = K << 4;   // 32*K/2
    for (int i = tid; i < tot2; i += 256) {
        float2 v = *(const float2*)(sF + (i << 1));
        v2h h;
        h[0] = (_Float16)v.x;
        h[1] = (_Float16)v.y;
        *(v2h*)(sH + (i << 1)) = h;
    }
    __syncthreads();

    const int wid  = tid >> 5;
    const int lane = tid & 31;
    const int ct   = blockIdx.y * 8 + wid;
    const int MoPad16 = (Mo + 15) >> 4;
    if (ct >= MoPad16) return;

    const int half = lane >> 4;
    const int n    = lane & 15;
    const int col  = (ct << 4) + n;
    const int KC   = K >> 5;

    const _Float16* a0base = sH + n * K + (half << 4);     // strip 0 (rows 0..15)
    const _Float16* a1base = a0base + (K << 4);            // strip 1 (rows 16..31)
    const _Float16* bbase  = Wp + ((size_t)(ct * KC) * 32 + lane) * 16;

    v8f c0, c1;
    if (acc) {
        #pragma unroll
        for (int r = 0; r < 8; ++r) {
            int row0 = rowbase + r + (half << 3);
            int row1 = row0 + 16;
            c0[r] = (col < Mo && row0 < M) ? C[(size_t)row0 * ldc + col] : 0.0f;
            c1[r] = (col < Mo && row1 < M) ? C[(size_t)row1 * ldc + col] : 0.0f;
        }
    } else {
        #pragma unroll
        for (int r = 0; r < 8; ++r) { c0[r] = 0.0f; c1[r] = 0.0f; }
    }

    for (int kc = 0; kc < KC; ++kc) {
        v16h b = *(const v16h*)(bbase + (size_t)kc * 512);   // 32 lanes * 16 halves
        int k0 = kc << 5;
        v8h a0lo = *(const v8h*)(a0base + k0);
        v8h a0hi = *(const v8h*)(a0base + k0 + 8);
        v8h a1lo = *(const v8h*)(a1base + k0);
        v8h a1hi = *(const v8h*)(a1base + k0 + 8);
        v16h a0, a1;
        #pragma unroll
        for (int j = 0; j < 8; ++j) {
            a0[j] = a0lo[j]; a0[j + 8] = a0hi[j];
            a1[j] = a1lo[j]; a1[j + 8] = a1hi[j];
        }
        c0 = __builtin_amdgcn_wmma_f32_16x16x32_f16(false, a0, false, b,
                                                    (short)0, c0, false, false);
        c1 = __builtin_amdgcn_wmma_f32_16x16x32_f16(false, a1, false, b,
                                                    (short)0, c1, false, false);
    }

    float alpha = pa ? pa[0] : 1.0f;
    int use_prelu = (pa != nullptr);
    float bv = (bias && col < Mo) ? bias[col] : 0.0f;
    #pragma unroll
    for (int r = 0; r < 8; ++r) {
        int row0 = rowbase + r + (half << 3);
        int row1 = row0 + 16;
        if (col < Mo && row0 < M) {
            float v = c0[r] + bv;
            if (use_prelu && v < 0.0f) v *= alpha;
            C[(size_t)row0 * ldc + col] = v;
        }
        if (col < Mo && row1 < M) {
            float v = c1[r] + bv;
            if (use_prelu && v < 0.0f) v *= alpha;
            C[(size_t)row1 * ldc + col] = v;
        }
    }
}

// ---------------------------------------------------------------------------
// GAT attention scalars
// ---------------------------------------------------------------------------
__global__ void att_scores(const float* __restrict__ h, const float* __restrict__ asrc,
                           const float* __restrict__ adst, float* __restrict__ as_out,
                           float* __restrict__ ad_out, int n) {
    int t = blockIdx.x * blockDim.x + threadIdx.x;
    if (t >= n * 2) return;
    int node = t >> 1, hd = t & 1;
    const float4* hp = (const float4*)(h + (size_t)node * 256 + hd * 128);
    const float4* ws = (const float4*)(asrc + hd * 128);
    const float4* wd = (const float4*)(adst + hd * 128);
    float ss = 0.0f, sd = 0.0f;
    #pragma unroll 4
    for (int i = 0; i < 32; ++i) {
        float4 hv = hp[i], s4 = ws[i], d4 = wd[i];
        ss += hv.x * s4.x + hv.y * s4.y + hv.z * s4.z + hv.w * s4.w;
        sd += hv.x * d4.x + hv.y * d4.y + hv.z * d4.z + hv.w * d4.w;
    }
    as_out[t] = ss;
    ad_out[t] = sd;
}

__device__ __forceinline__ unsigned ordkey(float f) {
    unsigned b = __float_as_uint(f);
    return (b & 0x80000000u) ? ~b : (b | 0x80000000u);
}
__device__ __forceinline__ float ordval(unsigned k) {
    unsigned b = (k & 0x80000000u) ? (k ^ 0x80000000u) : ~k;
    return __uint_as_float(b);
}

__global__ void edge_max(const int* __restrict__ src, const int* __restrict__ dst,
                         const float* __restrict__ as_, const float* __restrict__ ad_,
                         float* __restrict__ ebuf, unsigned* __restrict__ segmax,
                         int E, int n) {
    int e = blockIdx.x * blockDim.x + threadIdx.x;
    if (e >= E + n) return;
    int s = (e < E) ? src[e] : (e - E);
    int d = (e < E) ? dst[e] : (e - E);
    #pragma unroll
    for (int hd = 0; hd < 2; ++hd) {
        float v = as_[s * 2 + hd] + ad_[d * 2 + hd];
        v = (v >= 0.0f) ? v : 0.2f * v;
        ebuf[(size_t)e * 2 + hd] = v;
        atomicMax(&segmax[d * 2 + hd], ordkey(v));
    }
}

__global__ void edge_exp(const int* __restrict__ dst, float* __restrict__ ebuf,
                         const unsigned* __restrict__ segmax, float* __restrict__ denom,
                         int E, int n) {
    int e = blockIdx.x * blockDim.x + threadIdx.x;
    if (e >= E + n) return;
    int d = (e < E) ? dst[e] : (e - E);
    #pragma unroll
    for (int hd = 0; hd < 2; ++hd) {
        float mx = ordval(segmax[d * 2 + hd]);
        float ex = __expf(ebuf[(size_t)e * 2 + hd] - mx);
        ebuf[(size_t)e * 2 + hd] = ex;
        atomicAdd(&denom[d * 2 + hd], ex);
    }
}

__global__ void edge_agg(const int* __restrict__ src, const int* __restrict__ dst,
                         const float* __restrict__ h, const float* __restrict__ ebuf,
                         const float* __restrict__ denom, float* __restrict__ out,
                         int E, int n) {
    long long gid = (long long)blockIdx.x * blockDim.x + threadIdx.x;
    long long tot = (long long)(E + n) * 64;
    if (gid >= tot) return;
    int e  = (int)(gid >> 6);
    int q  = (int)(gid & 63);
    int c0 = q << 2;
    int s = (e < E) ? src[e] : (e - E);
    int d = (e < E) ? dst[e] : (e - E);
    int hd = c0 >> 7;
    float den = fmaxf(denom[d * 2 + hd], 1e-16f);
    float alpha = ebuf[(size_t)e * 2 + hd] / den;
    float4 hv = *(const float4*)(h + (size_t)s * 256 + c0);
    float* o = out + (size_t)d * 256 + c0;
    atomicAdd(o + 0, hv.x * alpha);
    atomicAdd(o + 1, hv.y * alpha);
    atomicAdd(o + 2, hv.z * alpha);
    atomicAdd(o + 3, hv.w * alpha);
}

__global__ void bias_prelu(float* __restrict__ x, const float* __restrict__ b,
                           const float* __restrict__ pa, long long n, int F) {
    long long i = (long long)blockIdx.x * blockDim.x + threadIdx.x;
    if (i >= n) return;
    float v = x[i] + b[(int)(i % F)];
    float a = pa[0];
    x[i] = (v >= 0.0f) ? v : a * v;
}

__global__ void edge_cnt(const int* __restrict__ dst, float* __restrict__ cnt, int E) {
    int e = blockIdx.x * blockDim.x + threadIdx.x;
    if (e >= E) return;
    atomicAdd(&cnt[dst[e]], 1.0f);
}

__global__ void edge_scatter(const int* __restrict__ src, const int* __restrict__ dst,
                             const float* __restrict__ x, float* __restrict__ agg,
                             int E, int F, int qshift) {
    long long gid = (long long)blockIdx.x * blockDim.x + threadIdx.x;
    long long tot = (long long)E << qshift;
    if (gid >= tot) return;
    int e  = (int)(gid >> qshift);
    int q  = (int)(gid & ((1 << qshift) - 1));
    int c0 = q << 2;
    int s = src[e], d = dst[e];
    float4 v = *(const float4*)(x + (size_t)s * F + c0);
    float* o = agg + (size_t)d * F + c0;
    atomicAdd(o + 0, v.x);
    atomicAdd(o + 1, v.y);
    atomicAdd(o + 2, v.z);
    atomicAdd(o + 3, v.w);
}

__global__ void mean_div(float* __restrict__ agg, const float* __restrict__ cnt,
                         long long n, int F) {
    long long i = (long long)blockIdx.x * blockDim.x + threadIdx.x;
    if (i >= n) return;
    agg[i] /= fmaxf(cnt[(int)(i / F)], 1.0f);
}

__global__ void fill_f32(float* p, float v, long long n) {
    long long i = (long long)blockIdx.x * blockDim.x + threadIdx.x;
    if (i < n) p[i] = v;
}
__global__ void fill_u32(unsigned* p, unsigned v, long long n) {
    long long i = (long long)blockIdx.x * blockDim.x + threadIdx.x;
    if (i < n) p[i] = v;
}

// ---------------------------------------------------------------------------
extern "C" void kernel_launch(void* const* d_in, const int* in_sizes, int n_in,
                              void* d_out, int out_size, void* d_ws, size_t ws_size,
                              hipStream_t stream) {
    const int N = in_sizes[0] / 64;       // 100000
    const int E = in_sizes[1] / 2;        // 800000

    const float* x       = (const float*)d_in[0];
    const int*   ei      = (const int*)d_in[1];
    const int*   src     = ei;
    const int*   dst     = ei + E;
    const float* g1_W    = (const float*)d_in[2];
    const float* g1_asrc = (const float*)d_in[3];
    const float* g1_adst = (const float*)d_in[4];
    const float* g1_b    = (const float*)d_in[5];
    const float* g2_W    = (const float*)d_in[6];
    const float* g2_asrc = (const float*)d_in[7];
    const float* g2_adst = (const float*)d_in[8];
    const float* g2_b    = (const float*)d_in[9];
    const float* s1_Wl   = (const float*)d_in[10];
    const float* s1_bl   = (const float*)d_in[11];
    const float* s1_Wr   = (const float*)d_in[12];
    const float* s2_Wl   = (const float*)d_in[13];
    const float* s2_bl   = (const float*)d_in[14];
    const float* s2_Wr   = (const float*)d_in[15];
    const float* fus_W   = (const float*)d_in[16];
    const float* fus_b   = (const float*)d_in[17];
    const float* lvl_W1  = (const float*)d_in[18];
    const float* lvl_b1  = (const float*)d_in[19];
    const float* lvl_W2  = (const float*)d_in[20];
    const float* lvl_b2  = (const float*)d_in[21];
    const float* lvl_W3  = (const float*)d_in[22];
    const float* lvl_b3  = (const float*)d_in[23];
    const float* rock_W1 = (const float*)d_in[24];
    const float* rock_b1 = (const float*)d_in[25];
    const float* rock_W2 = (const float*)d_in[26];
    const float* rock_b2 = (const float*)d_in[27];
    const float* rock_W3 = (const float*)d_in[28];
    const float* rock_b3 = (const float*)d_in[29];
    const float* act_a   = (const float*)d_in[30];
    const float* lvl_a1  = (const float*)d_in[31];
    const float* lvl_a2  = (const float*)d_in[32];
    const float* rock_a1 = (const float*)d_in[33];
    const float* rock_a2 = (const float*)d_in[34];

    float* wsf = (float*)d_ws;
    float*    A   = wsf;                          // N*256
    float*    B   = A + (size_t)N * 256;          // N*256
    float*    Cb  = B + (size_t)N * 256;          // N*128
    float*    Db  = Cb + (size_t)N * 128;         // N*128
    float*    AS_ = Db + (size_t)N * 128;         // N*2
    float*    AD_ = AS_ + (size_t)N * 2;          // N*2
    unsigned* SM  = (unsigned*)(AD_ + (size_t)N * 2);   // N*2
    float*    DEN = (float*)SM + (size_t)N * 2;   // N*2
    float*    CNT = DEN + (size_t)N * 2;          // N
    float*    EB  = CNT + N;                      // (E+N)*2
    _Float16* WP  = (_Float16*)(EB + (size_t)(E + N) * 2);

    _Float16* p_g1W  = WP;
    _Float16* p_g2W  = p_g1W  + 64 * 256;
    _Float16* p_s1Wl = p_g2W  + 256 * 256;
    _Float16* p_s1Wr = p_s1Wl + 256 * 128;
    _Float16* p_s2Wl = p_s1Wr + 256 * 128;
    _Float16* p_s2Wr = p_s2Wl + 128 * 128;
    _Float16* p_fus1 = p_s2Wr + 128 * 128;
    _Float16* p_fus2 = p_fus1 + 256 * 128;
    _Float16* p_lW1  = p_fus2 + 128 * 128;
    _Float16* p_lW2  = p_lW1  + 128 * 64;
    _Float16* p_lW3  = p_lW2  + 64 * 32;
    _Float16* p_rW1  = p_lW3  + 32 * 16;
    _Float16* p_rW2  = p_rW1  + 128 * 64;
    _Float16* p_rW3  = p_rW2  + 64 * 32;

    auto pack = [&](const float* W, _Float16* P, int K, int Mo, int MoPad) {
        int total = (MoPad / 16) * (K / 32) * 32;
        pack_w<<<(total + 255) / 256, 256, 0, stream>>>(W, P, K, Mo, MoPad);
    };
    auto gemm = [&](const float* Ain, const _Float16* Wp, float* Co,
                    const float* bias, const float* pa,
                    int K, int Mo, int ldc, int acc) {
        int MoPad16 = (Mo + 15) / 16;
        dim3 grid((N + 31) / 32, (MoPad16 + 7) / 8);
        size_t shmem = (size_t)32 * K * 6;   // f32 tile + f16 tile
        wmma_gemm<<<grid, 256, shmem, stream>>>(Ain, Wp, Co, bias, pa, N, K, Mo, ldc, acc);
    };
    auto fillf = [&](float* p, float v, long long c) {
        fill_f32<<<(unsigned)((c + 255) / 256), 256, 0, stream>>>(p, v, c);
    };
    auto fillu = [&](unsigned* p, unsigned v, long long c) {
        fill_u32<<<(unsigned)((c + 255) / 256), 256, 0, stream>>>(p, v, c);
    };

    pack(g1_W, p_g1W, 64, 256, 256);
    pack(g2_W, p_g2W, 256, 256, 256);
    pack(s1_Wl, p_s1Wl, 256, 128, 128);
    pack(s1_Wr, p_s1Wr, 256, 128, 128);
    pack(s2_Wl, p_s2Wl, 128, 128, 128);
    pack(s2_Wr, p_s2Wr, 128, 128, 128);
    pack(fus_W, p_fus1, 256, 128, 128);
    pack(fus_W + 256 * 128, p_fus2, 128, 128, 128);
    pack(lvl_W1, p_lW1, 128, 64, 64);
    pack(lvl_W2, p_lW2, 64, 32, 32);
    pack(lvl_W3, p_lW3, 32, 1, 16);
    pack(rock_W1, p_rW1, 128, 64, 64);
    pack(rock_W2, p_rW2, 64, 32, 32);
    pack(rock_W3, p_rW3, 32, 13, 16);

    const int EN = E + N;
    const int ebks = (EN + 255) / 256;
    const unsigned agbks = (unsigned)(((long long)EN * 64 + 255) / 256);

    // ---- GAT layer 1 ----
    gemm(x, p_g1W, A, nullptr, nullptr, 64, 256, 256, 0);
    att_scores<<<(2 * N + 255) / 256, 256, 0, stream>>>(A, g1_asrc, g1_adst, AS_, AD_, N);
    fillu(SM, 0u, 2LL * N);
    fillf(DEN, 0.0f, 2LL * N);
    fillf(B, 0.0f, (long long)N * 256);
    edge_max<<<ebks, 256, 0, stream>>>(src, dst, AS_, AD_, EB, SM, E, N);
    edge_exp<<<ebks, 256, 0, stream>>>(dst, EB, SM, DEN, E, N);
    edge_agg<<<agbks, 256, 0, stream>>>(src, dst, A, EB, DEN, B, E, N);
    bias_prelu<<<(unsigned)(((long long)N * 256 + 255) / 256), 256, 0, stream>>>(
        B, g1_b, act_a, (long long)N * 256, 256);

    // ---- GAT layer 2 ----
    gemm(B, p_g2W, A, nullptr, nullptr, 256, 256, 256, 0);
    att_scores<<<(2 * N + 255) / 256, 256, 0, stream>>>(A, g2_asrc, g2_adst, AS_, AD_, N);
    fillu(SM, 0u, 2LL * N);
    fillf(DEN, 0.0f, 2LL * N);
    fillf(B, 0.0f, (long long)N * 256);
    edge_max<<<ebks, 256, 0, stream>>>(src, dst, AS_, AD_, EB, SM, E, N);
    edge_exp<<<ebks, 256, 0, stream>>>(dst, EB, SM, DEN, E, N);
    edge_agg<<<agbks, 256, 0, stream>>>(src, dst, A, EB, DEN, B, E, N);
    bias_prelu<<<(unsigned)(((long long)N * 256 + 255) / 256), 256, 0, stream>>>(
        B, g2_b, act_a, (long long)N * 256, 256);

    // ---- counts ----
    fillf(CNT, 0.0f, N);
    edge_cnt<<<(E + 255) / 256, 256, 0, stream>>>(dst, CNT, E);

    // ---- SAGE layer 1 ----
    fillf(A, 0.0f, (long long)N * 256);
    edge_scatter<<<(unsigned)(((long long)E * 64 + 255) / 256), 256, 0, stream>>>(
        src, dst, B, A, E, 256, 6);
    mean_div<<<(unsigned)(((long long)N * 256 + 255) / 256), 256, 0, stream>>>(
        A, CNT, (long long)N * 256, 256);
    gemm(A, p_s1Wl, Cb, nullptr, nullptr, 256, 128, 128, 0);
    gemm(B, p_s1Wr, Cb, s1_bl, act_a, 256, 128, 128, 1);

    // ---- SAGE layer 2 ----
    fillf(A, 0.0f, (long long)N * 128);
    edge_scatter<<<(unsigned)(((long long)E * 32 + 255) / 256), 256, 0, stream>>>(
        src, dst, Cb, A, E, 128, 5);
    mean_div<<<(unsigned)(((long long)N * 128 + 255) / 256), 256, 0, stream>>>(
        A, CNT, (long long)N * 128, 128);
    gemm(A, p_s2Wl, Db, nullptr, nullptr, 128, 128, 128, 0);
    gemm(Cb, p_s2Wr, Db, s2_bl, act_a, 128, 128, 128, 1);

    // ---- fusion ----
    gemm(B, p_fus1, Cb, nullptr, nullptr, 256, 128, 128, 0);
    gemm(Db, p_fus2, Cb, fus_b, act_a, 128, 128, 128, 1);

    // ---- heads ----
    float* out = (float*)d_out;
    gemm(Cb, p_lW1, A, lvl_b1, lvl_a1, 128, 64, 64, 0);
    gemm(A, p_lW2, Db, lvl_b2, lvl_a2, 64, 32, 32, 0);
    gemm(Db, p_lW3, out, lvl_b3, nullptr, 32, 1, 1, 0);

    gemm(Cb, p_rW1, A, rock_b1, rock_a1, 128, 64, 64, 0);
    gemm(A, p_rW2, Db, rock_b2, rock_a2, 64, 32, 32, 0);
    gemm(Db, p_rW3, out + N, rock_b3, nullptr, 32, 13, 13, 0);
}
